// PureRNN_69209103007832
// MI455X (gfx1250) — matrix operand order repaired and checked by reading, not compiled, not run on hardware
//
#include <hip/hip_runtime.h>

// ---------------------------------------------------------------------------
// 4-layer LSTM, B=64, T=256, H=1024 on MI455X (gfx1250).
// Persistent kernel: bf16 weights + x live in L2 (~101MB < 192MB L2),
// device-wide barrier between the 1024 sequential (layer,timestep) cells,
// v_wmma_f32_16x16x32_bf16 with f32 accumulation for both GEMMs of a cell.
// All GEMM A/B sources are bf16 (x pre-converted in the parallel prep pass)
// so the inner loop is branch-free and WMMA always runs with EXEC all-ones.
// ---------------------------------------------------------------------------

#define NL   4
#define NB   64
#define NT   256
#define NH   1024
#define NG   (4 * NH)          // 4096 gate columns
#define GRID_WGS 64            // one WG per 16 hidden columns

typedef __attribute__((ext_vector_type(16))) __bf16         v16bf;
typedef __attribute__((ext_vector_type(8)))  float          v8f;
typedef __attribute__((ext_vector_type(16))) unsigned short v16us;
typedef __attribute__((ext_vector_type(8)))  unsigned short v8us;

__device__ __forceinline__ unsigned short f2bf(float f) {
    unsigned u = __float_as_uint(f);
    u += 0x7FFFu + ((u >> 16) & 1u);       // round-to-nearest-even
    return (unsigned short)(u >> 16);
}

// ---------------------------------------------------------------------------
// Prep (fully parallel, off the critical path):
//  - W[l][n][k] fp32 -> Wt[l][k][n] bf16 ([K][N] so a WMMA B-fragment is
//    lane = N column, 16 K values strided by 4H, lanes coalesced along N)
//  - x fp32 -> x_bf bf16 (same [B,T,H] layout; element count matches weights)
// ---------------------------------------------------------------------------
__global__ void prep_weights(const float* __restrict__ Wih,
                             const float* __restrict__ Whh,
                             const float* __restrict__ x,
                             unsigned short* __restrict__ Wt_ih,
                             unsigned short* __restrict__ Wt_hh,
                             unsigned short* __restrict__ x_bf) {
    size_t id = (size_t)blockIdx.x * blockDim.x + threadIdx.x; // over NL*NH*NG
    int n = (int)(id % NG);
    size_t rem = id / NG;
    int k = (int)(rem % NH);
    int l = (int)(rem / NH);
    size_t src = ((size_t)l * NG + n) * NH + k;
    Wt_ih[id] = f2bf(Wih[src]);
    Wt_hh[id] = f2bf(Whh[src]);
    x_bf[id]  = f2bf(x[id]);               // NL*NH*NG == NB*NT*NH
}

// Fold biases, zero h/c state and the grid barrier (ws is poisoned 0xAA).
__global__ void prep_state(const float* __restrict__ bih,
                           const float* __restrict__ bhh,
                           float* __restrict__ bias,
                           unsigned short* __restrict__ hbf,
                           float* __restrict__ c,
                           unsigned int* __restrict__ bar) {
    int id = blockIdx.x * blockDim.x + threadIdx.x;
    if (id < NL * NG) bias[id] = bih[id] + bhh[id];
    if (id < NL * NB * NH) { hbf[id] = 0; c[id] = 0.0f; }
    if (id < 2) bar[id] = 0;
}

// ---------------------------------------------------------------------------
// Device-wide sense-reversing barrier (64 co-resident WGs).
// ---------------------------------------------------------------------------
__device__ __forceinline__ void grid_barrier(unsigned int* cnt, unsigned int* gen,
                                             unsigned int nwg) {
    __syncthreads();
    if (threadIdx.x == 0) {
        __threadfence();
        unsigned int my_gen  = __atomic_load_n(gen, __ATOMIC_RELAXED);
        unsigned int arrived = __atomic_add_fetch(cnt, 1u, __ATOMIC_ACQ_REL);
        if (arrived == nwg) {
            __atomic_store_n(cnt, 0u, __ATOMIC_RELAXED);
            __atomic_fetch_add(gen, 1u, __ATOMIC_RELEASE);
        } else {
            while (__atomic_load_n(gen, __ATOMIC_ACQUIRE) == my_gen)
                __builtin_amdgcn_s_sleep(1);
        }
        __threadfence();
    }
    __syncthreads();
}

// ---------------------------------------------------------------------------
// Persistent LSTM kernel. Grid: 64 WGs x 256 threads (8 wave32).
//   WG w owns hidden columns [w*16, w*16+16).
//   wave = khalf*4 + g : gate g in {i,f,g,o}; khalf 0 = input GEMM (Wt_ih),
//   khalf 1 = recurrent GEMM (Wt_hh). Each wave: 4 M-tiles (B=64) x 1 N-tile,
//   K=1024 in 32 steps of v_wmma_f32_16x16x32_bf16; the B-fragment is loaded
//   once per K step and reused by all 4 accumulators. Partials merge in LDS,
//   then the LSTM pointwise math runs WG-locally.
// ---------------------------------------------------------------------------
__launch_bounds__(256, 1)
__global__ void lstm_persistent(const unsigned short* __restrict__ x_bf, // [B,T,H] bf16
                                const unsigned short* __restrict__ Wt_ih, // [L][K=H][N=4H]
                                const unsigned short* __restrict__ Wt_hh,
                                const float* __restrict__ bias,       // [L][4H]
                                unsigned short* __restrict__ hbf,     // [L][B][H] bf16
                                float* __restrict__ c,                // [L][B][H] f32
                                float* __restrict__ out,              // [B,H] f32
                                unsigned int* __restrict__ bar) {
    __shared__ float lds_g[4][NB][16];

    const int tid   = threadIdx.x;
    const int lane  = tid & 31;
    const int wave  = tid >> 5;
    const int g     = wave & 3;            // gate index
    const int khalf = wave >> 2;           // 0: input GEMM, 1: recurrent GEMM
    const int j0    = blockIdx.x * 16;     // hidden-column base of this WG

    const int arow  = lane & 15;           // M within a 16-row tile (A layout)
    const int ksub  = (lane >> 4) * 8;     // K sub-slice select (A layout)
    const int dcol  = lane & 15;           // N for C/D layout
    const int drow  = (lane >> 4) * 8;     // M offset for C/D layout

    for (int t = 0; t < NT; ++t) {
        for (int l = 0; l < NL; ++l) {
            // ---------------- GEMM phase (branch-free, all bf16) ----------
            v8f acc[4] = {v8f{}, v8f{}, v8f{}, v8f{}};

            const unsigned short* W =
                (khalf == 0 ? Wt_ih : Wt_hh) + (size_t)l * NH * NG + (size_t)(g * NH + j0);

            // A source: layer-0 input waves read x_bf[:,t,:] (row stride T*H),
            // everything else reads h state (row stride H). Pure pointer math,
            // no control flow -> EXEC stays all-ones around the WMMAs.
            const bool   use_x   = (khalf == 0) && (l == 0);
            const int    hl      = (khalf == 0) ? (l > 0 ? l - 1 : 0) : l;
            const size_t astride = use_x ? (size_t)NT * NH : (size_t)NH;
            const unsigned short* A0 =
                use_x ? (x_bf + (size_t)t * NH) : (hbf + (size_t)hl * NB * NH);
            const unsigned short* Arow = A0 + (size_t)arow * astride + ksub;

#pragma unroll 2
            for (int k0 = 0; k0 < NH; k0 += 32) {
                // B fragment: lane holds N column (lane&15), 16 K values in table order.
                const unsigned short* wp = W + (size_t)(k0 + lane) * NG;
                const unsigned short* wc = W + (size_t)k0 * NG + dcol;
                v16us braw;
#pragma unroll
                for (int s = 0; s < 8; ++s) {
                    braw[s]     = wc[(size_t)(ksub + s) * NG];
                    braw[s + 8] = wc[(size_t)(16 + ksub + s) * NG];
                }
                v16bf bfrag = __builtin_bit_cast(v16bf, braw);
                __builtin_prefetch(wp + (size_t)32 * NG, 0, 1); // next K block (L2)

#pragma unroll
                for (int m = 0; m < 4; ++m) {
                    const unsigned short* ap = Arow + (size_t)(m * 16) * astride + k0;
                    v8us lo = *(const v8us*)ap;
                    v8us hi = *(const v8us*)(ap + 16);
                    v16us araw;
#pragma unroll
                    for (int i = 0; i < 8; ++i) { araw[i] = lo[i]; araw[i + 8] = hi[i]; }
                    v16bf afrag = __builtin_bit_cast(v16bf, araw);
                    acc[m] = __builtin_amdgcn_wmma_f32_16x16x32_bf16(
                        false, afrag, false, bfrag, (short)0, acc[m], false, false);
                }
            }

            // ---------------- merge partials in LDS ----------------
            if (khalf == 0) {
#pragma unroll
                for (int m = 0; m < 4; ++m)
#pragma unroll
                    for (int r = 0; r < 8; ++r)
                        lds_g[g][m * 16 + drow + r][dcol] = acc[m][r];
            }
            __syncthreads();
            if (khalf == 1) {
#pragma unroll
                for (int m = 0; m < 4; ++m)
#pragma unroll
                    for (int r = 0; r < 8; ++r)
                        lds_g[g][m * 16 + drow + r][dcol] += acc[m][r];
            }
            __syncthreads();

            // ---------------- LSTM cell pointwise ----------------
            const float* bias_l = bias + l * NG;
            float* c_l = c + (size_t)l * NB * NH;
            unsigned short* h_l = hbf + (size_t)l * NB * NH;
            const bool last = (l == NL - 1) && (t == NT - 1);
#pragma unroll
            for (int e = 0; e < 4; ++e) {
                const int p = tid + 256 * e;        // 1024 (b,j) pairs / 256 threads
                const int b = p >> 4;
                const int j = p & 15;
                const int ja = j0 + j;
                const float iv = lds_g[0][b][j] + bias_l[0 * NH + ja];
                const float fv = lds_g[1][b][j] + bias_l[1 * NH + ja];
                const float gv = lds_g[2][b][j] + bias_l[2 * NH + ja];
                const float ov = lds_g[3][b][j] + bias_l[3 * NH + ja];
                const float is = 1.0f / (1.0f + __expf(-iv));
                const float fs = 1.0f / (1.0f + __expf(-fv));
                const float os = 1.0f / (1.0f + __expf(-ov));
                const float gt = tanhf(gv);
                const size_t ci = (size_t)b * NH + ja;
                const float cn = fs * c_l[ci] + is * gt;
                const float hn = os * tanhf(cn);
                c_l[ci] = cn;
                h_l[ci] = f2bf(hn);
                if (last) out[ci] = hn;
            }

            // make h/c visible to all WGs before the next dependent cell
            grid_barrier(bar, bar + 1, (unsigned)gridDim.x);
        }
    }
}

// ---------------------------------------------------------------------------
extern "C" void kernel_launch(void* const* d_in, const int* in_sizes, int n_in,
                              void* d_out, int out_size, void* d_ws, size_t ws_size,
                              hipStream_t stream) {
    (void)in_sizes; (void)n_in; (void)out_size; (void)ws_size;
    const float* x   = (const float*)d_in[0];
    const float* Wih = (const float*)d_in[1];
    const float* Whh = (const float*)d_in[2];
    const float* bih = (const float*)d_in[3];
    const float* bhh = (const float*)d_in[4];

    // Workspace layout (~102.3 MB, L2-resident working set):
    char* ws = (char*)d_ws;
    const size_t WB = (size_t)NL * NH * NG * sizeof(unsigned short); // 33,554,432 each
    unsigned short* Wt_ih = (unsigned short*)(ws);
    unsigned short* Wt_hh = (unsigned short*)(ws + WB);
    unsigned short* x_bf  = (unsigned short*)(ws + 2 * WB);
    float*          bias  = (float*)         (ws + 3 * WB);
    unsigned short* hbf   = (unsigned short*)(ws + 3 * WB + 65536);
    float*          cbuf  = (float*)         (ws + 3 * WB + 65536 + 524288);
    unsigned int*   bar   = (unsigned int*)  (ws + 3 * WB + 65536 + 524288 + 1048576);

    const int welems = NL * NH * NG;                       // 16,777,216
    prep_weights<<<welems / 256, 256, 0, stream>>>(Wih, Whh, x, Wt_ih, Wt_hh, x_bf);
    prep_state<<<(NL * NB * NH) / 256, 256, 0, stream>>>(bih, bhh, bias, hbf, cbuf, bar);
    lstm_persistent<<<GRID_WGS, 256, 0, stream>>>(x_bf, Wt_ih, Wt_hh, bias, hbf, cbuf,
                                                  (float*)d_out, bar);
}
